// LocalEnergyCore_29265907155499
// MI455X (gfx1250) — matrix operand order repaired and checked by reading, not compiled, not run on hardware
//
#include <hip/hip_runtime.h>

typedef _Float16 h8   __attribute__((ext_vector_type(8)));
typedef _Float16 v16h __attribute__((ext_vector_type(16)));
typedef float    v8f  __attribute__((ext_vector_type(8)));

#define B_N   4096
#define K_CH  8
#define HDIM  64
#define WDIM  64
#define S_N   50
#define HID   64
#define CTXN  71           // 9*K - 1
#define KPAD  96           // 3 x 32 (f16 WMMA K-steps)
#define ROWS  (B_N * S_N)  // 204800 == 16 * 12800 exactly
#define ROW_TILES (ROWS / 16)

// ---- workspace layout ----
// [0..3]      int      mismatch counter
// [256..]     _Float16 packed W1 B-fragments: [ntile(4)][kstep(3)][lane(32)][16]
#define WS_BPACK_OFF 256
#define BPACK_ELEMS  (4 * 3 * 32 * 16)   // 6144 halves = 12 KB

// Pack W1 (71x64 f32) into f16 WMMA B-fragments; B 32x16 layout (matches A):
// lanes 0-15 hold column N=lane, K = kstep*32 + {0..7,16..23} ; lanes 16-31 hold
// N=lane-16, K = kstep*32 + {8..15,24..31}.
__global__ void prep_kernel(const float* __restrict__ W1, int* counter,
                            _Float16* __restrict__ bpack) {
    int t = blockIdx.x * blockDim.x + threadIdx.x;
    if (t == 0) *counter = 0;
    if (t >= BPACK_ELEMS) return;
    int h    = t & 15;
    int lane = (t >> 4) & 31;
    int rest = t >> 9;          // nt*3 + kk
    int kk   = rest % 3;
    int nt   = rest / 3;
    int col  = nt * 16 + (lane & 15);
    int k    = kk * 32 + (h & 8) * 2 + (lane >> 4) * 8 + (h & 7);
    bpack[t] = (k < CTXN) ? (_Float16)W1[k * HID + col] : (_Float16)0.0f;
}

__global__ __launch_bounds__(128)
void energy_kernel(const float* __restrict__ z,
                   const float* __restrict__ b1,
                   const float* __restrict__ W2,
                   const float* __restrict__ b2,
                   const int*   __restrict__ b_idx,
                   const int*   __restrict__ i_idx,
                   const int*   __restrict__ j_idx,
                   const _Float16* __restrict__ bpack,
                   int* __restrict__ counter) {
    __shared__ _Float16 sA[16 * KPAD];   // 16 rows x 96 f16 = 3 KB
    __shared__ float    sPart[4 * 16];   // per-wave row partials

    const int tid  = threadIdx.x;
    const int r0   = blockIdx.x * 16;

    // ---- gather: 16 rows x 71 ctx features (pad K to 96 with zeros) ----
    for (int e = tid; e < 16 * KPAD; e += 128) {      // uniform trip count
        int row = e / KPAD;
        int c   = e - row * KPAD;
        float v = 0.0f;
        if (c < CTXN) {
            int r  = r0 + row;
            int b  = r / S_N;
            int s  = r - b * S_N;
            int bq = b_idx[s];
            int ii = i_idx[s];
            int jj = j_idx[s];
            int cp = c + (c >= (4 * K_CH + bq));      // skip dropped center feat
            int t9 = cp >> 3;                          // neighbor 0..8
            int kc = cp & 7;                           // channel
            int ni = (ii + (t9 / 3) - 1 + HDIM) & (HDIM - 1);
            int nj = (jj + (t9 % 3) - 1 + WDIM) & (WDIM - 1);
            v = z[((b * K_CH + kc) << 12) + (ni << 6) + nj];
        }
        sA[e] = (_Float16)v;
    }
    __syncthreads();

    const int lane = tid & 31;
    const int wv   = tid >> 5;     // N-tile 0..3 (16 hidden cols each)
    const int lrow = lane & 15;
    const int lhi  = lane >> 4;

    // ---- A fragments (16x32 f16 / K-step), documented ISA layout:
    // lanes 0-15: M=lane, K=base+{0..7,16..23}; lanes 16-31: M=lane-16,
    // K=base+{8..15,24..31}. Each half is 8 contiguous f16 in LDS (b128).
    const _Float16* rp = &sA[lrow * KPAD];
    v16h a0, a1, a2;
    #pragma unroll
    for (int kk = 0; kk < 3; kk++) {
        int kb = kk * 32 + lhi * 8;
        h8 lo = *(const h8*)(rp + kb);
        h8 hi = *(const h8*)(rp + kb + 16);
        v16h av;
        #pragma unroll
        for (int h = 0; h < 8; h++) { av[h] = lo[h]; av[8 + h] = hi[h]; }
        if (kk == 0) a0 = av; else if (kk == 1) a1 = av; else a2 = av;
    }

    // ---- B fragments: prepacked, 32B contiguous per lane ----
    const _Float16* bp = bpack + ((wv * 3) * 32 + lane) * 16;
    v16h bf0 = *(const v16h*)(bp);
    v16h bf1 = *(const v16h*)(bp + 32 * 16);
    v16h bf2 = *(const v16h*)(bp + 2 * 32 * 16);

    // ---- 3 chained WMMAs: h-tile (16 rows x 16 hidden) in f32 ----
    v8f acc = {};
    acc = __builtin_amdgcn_wmma_f32_16x16x32_f16(false, a0, false, bf0,
                                                 (short)0, acc, false, false);
    acc = __builtin_amdgcn_wmma_f32_16x16x32_f16(false, a1, false, bf1,
                                                 (short)0, acc, false, false);
    acc = __builtin_amdgcn_wmma_f32_16x16x32_f16(false, a2, false, bf2,
                                                 (short)0, acc, false, false);

    // ---- epilogue: relu(h+b1)*W2, reduce 16 cols across lanes ----
    // C layout: lane L, VGPR j -> row = j + 8*(L>=16), col = L%16.
    int   col   = wv * 16 + lrow;
    float biasv = b1[col];
    float w2v   = W2[col];
    float p[8];
    #pragma unroll
    for (int jr = 0; jr < 8; jr++) {
        float hv = acc[jr] + biasv;
        hv = hv > 0.0f ? hv : 0.0f;
        p[jr] = hv * w2v;
    }
    #pragma unroll
    for (int off = 1; off < 16; off <<= 1) {   // stays within each 16-lane half
        #pragma unroll
        for (int jr = 0; jr < 8; jr++)
            p[jr] += __shfl_xor(p[jr], off, 32);
    }
    if (lrow == 0) {   // lane 0 -> rows 0..7 ; lane 16 -> rows 8..15
        #pragma unroll
        for (int jr = 0; jr < 8; jr++)
            sPart[wv * 16 + lhi * 8 + jr] = p[jr];
    }
    __syncthreads();

    // ---- combine 4 waves, threshold, compare with target, count ----
    int mis = 0;
    if (tid < 16) {
        float logit = sPart[tid] + sPart[16 + tid] + sPart[32 + tid]
                    + sPart[48 + tid] + b2[0];
        int r  = r0 + tid;
        int b  = r / S_N;
        int s  = r - b * S_N;
        int bq = b_idx[s];
        float target = z[((b * K_CH + bq) << 12) + (i_idx[s] << 6) + j_idx[s]];
        float pred   = (logit > 0.0f) ? 1.0f : 0.0f;
        mis = (pred != target) ? 1 : 0;
    }
    if (tid < 32) {
        unsigned long long m = __ballot(mis);
        if (tid == 0 && m) atomicAdd(counter, (int)__popcll(m));
    }
}

__global__ void finalize_kernel(const int* __restrict__ counter,
                                float* __restrict__ out) {
    if (threadIdx.x == 0 && blockIdx.x == 0)
        out[0] = (float)(*counter) * (1.0f / (float)ROWS);
}

extern "C" void kernel_launch(void* const* d_in, const int* in_sizes, int n_in,
                              void* d_out, int out_size, void* d_ws, size_t ws_size,
                              hipStream_t stream) {
    const float* z     = (const float*)d_in[0];
    const float* W1    = (const float*)d_in[1];
    const float* b1    = (const float*)d_in[2];
    const float* W2    = (const float*)d_in[3];
    const float* b2    = (const float*)d_in[4];
    const int*   b_idx = (const int*)d_in[5];
    const int*   i_idx = (const int*)d_in[6];
    const int*   j_idx = (const int*)d_in[7];

    int*      counter = (int*)d_ws;
    _Float16* bpack   = (_Float16*)((char*)d_ws + WS_BPACK_OFF);

    prep_kernel<<<(BPACK_ELEMS + 255) / 256, 256, 0, stream>>>(W1, counter, bpack);
    energy_kernel<<<ROW_TILES, 128, 0, stream>>>(z, b1, W2, b2,
                                                 b_idx, i_idx, j_idx,
                                                 bpack, counter);
    finalize_kernel<<<1, 32, 0, stream>>>(counter, (float*)d_out);
}
